// PointNet_SA_Module_70549132804475
// MI455X (gfx1250) — matrix operand order repaired and checked by reading, not compiled, not run on hardware
//
#include <hip/hip_runtime.h>
#include <hip/hip_bf16.h>
#include <stdint.h>

// ---------------- problem constants ----------------
#define B_    8
#define N_    8192
#define C_    64
#define M_    2048      // NPOINT
#define K_    32        // NSAMPLE
#define COLS  (B_*M_*K_)   // 524288 GEMM columns (BN stats span all of them)
#define CIN1  67
#define CP1   96        // channel pad for layer 1 (3 k-steps of 32)
#define O1    64
#define O2    64
#define O3    128
#define RAD2  0.04f
#define EPS_  1e-5f

typedef __attribute__((ext_vector_type(16))) __bf16 bf16x16;
typedef __attribute__((ext_vector_type(8)))  float  f32x8;
typedef unsigned int u32x4 __attribute__((ext_vector_type(4)));
typedef unsigned int u32x8 __attribute__((ext_vector_type(8)));

union FragBF { bf16x16 v; uint4 q[2]; };

__device__ __forceinline__ float bf2f(unsigned short h) {
    union { unsigned int u; float f; } x; x.u = ((unsigned int)h) << 16; return x.f;
}
__device__ __forceinline__ unsigned short f2bf(float f) {
    union { float f; unsigned int u; } x; x.f = f;
    unsigned int u = x.u;
    return (unsigned short)((u + 0x7fffu + ((u >> 16) & 1u)) >> 16);  // RNE
}
__device__ __forceinline__ unsigned lds_off(const void* p) {
    return (unsigned)(uintptr_t)p;   // low 32b of generic LDS pointer = LDS byte offset
}

// ---------------- 1) farthest point sampling (one WG per batch) ----------------
__global__ void fps_kernel(const float* __restrict__ xyz, float* __restrict__ out_newxyz)
{
    __shared__ float dist[N_];        // 32 KB
    __shared__ float redV[256];
    __shared__ int   redI[256];
    __shared__ float cx, cy, cz;
    __shared__ int   sFar;

    const int b = blockIdx.x, t = threadIdx.x;
    const float* px = xyz + (size_t)b * 3 * N_;
    const float* py = px + N_;
    const float* pz = px + 2 * N_;

    for (int i = t; i < N_; i += 256) dist[i] = 1e10f;
    if (t == 0) sFar = 0;
    __syncthreads();

    for (int it = 0; it < M_; ++it) {
        if (t == 0) {
            int far = sFar;
            float fx = px[far], fy = py[far], fz = pz[far];
            cx = fx; cy = fy; cz = fz;
            out_newxyz[(size_t)b*3*M_ + 0*M_ + it] = fx;   // new_xyz (B,3,M)
            out_newxyz[(size_t)b*3*M_ + 1*M_ + it] = fy;
            out_newxyz[(size_t)b*3*M_ + 2*M_ + it] = fz;
        }
        __syncthreads();
        float bx = cx, by = cy, bz = cz;
        float bestV = -1.f; int bestI = 0;
        for (int n = t; n < N_; n += 256) {
            float dx = px[n]-bx, dy = py[n]-by, dz = pz[n]-bz;
            float d  = dx*dx + dy*dy + dz*dz;
            float dd = dist[n]; dd = d < dd ? d : dd; dist[n] = dd;
            if (dd > bestV) { bestV = dd; bestI = n; }   // first max wins (ascending n)
        }
        redV[t] = bestV; redI[t] = bestI;
        __syncthreads();
        for (int s = 128; s > 0; s >>= 1) {
            if (t < s) {
                float v2 = redV[t+s]; int i2 = redI[t+s];
                if (v2 > redV[t] || (v2 == redV[t] && i2 < redI[t])) { redV[t] = v2; redI[t] = i2; }
            }
            __syncthreads();
        }
        if (t == 0) sFar = redI[0];
        __syncthreads();
    }
}

// ---------------- 2) ball query: one wave32 per centroid ----------------
__global__ void ballq_kernel(const float* __restrict__ xyz,
                             const float* __restrict__ newxyz,
                             int* __restrict__ gidx)
{
    __shared__ int shidx[8][K_];
    const int t = threadIdx.x, wave = t >> 5, lane = t & 31;
    const int g = blockIdx.x * 8 + wave;
    const int b = g / M_, m = g % M_;

    const float* px = xyz + (size_t)b * 3 * N_;
    const float* py = px + N_;
    const float* pz = px + 2 * N_;
    const float cx = newxyz[(size_t)b*3*M_ + 0*M_ + m];
    const float cy = newxyz[(size_t)b*3*M_ + 1*M_ + m];
    const float cz = newxyz[(size_t)b*3*M_ + 2*M_ + m];

    int cnt = 0;
    for (int base = 0; base < N_ && cnt < K_; base += 32) {
        int n = base + lane;
        float dx = px[n]-cx, dy = py[n]-cy, dz = pz[n]-cz;
        bool inr = (dx*dx + dy*dy + dz*dz) < RAD2;
        unsigned mask = (unsigned)__ballot(inr);          // wave32: low 32 bits
        int before = __popc(mask & ((1u << lane) - 1u));
        if (inr && (cnt + before) < K_) shidx[wave][cnt + before] = n;
        cnt += __popc(mask);
    }
    asm volatile("s_wait_dscnt 0" ::: "memory");          // own-wave LDS writes visible
    int first = (cnt > 0) ? shidx[wave][0] : 0;
    int v = (lane < cnt) ? shidx[wave][lane] : first;
    gidx[(size_t)g * K_ + lane] = v;
}

// ---------------- 3) gather + recentre + concat -> X1 bf16 [CP1][COLS] ----------------
__global__ void gather_kernel(const float* __restrict__ xyz,
                              const float* __restrict__ pts,
                              const float* __restrict__ newxyz,
                              const int* __restrict__ gidx,
                              unsigned short* __restrict__ X1)
{
    const int col = blockIdx.x * 256 + threadIdx.x;       // (b,m,k) flattened
    const int b = col / (M_*K_);
    const int rem = col % (M_*K_);
    const int m = rem / K_;
    const int n = gidx[col];

    const float* xb = xyz    + (size_t)b * 3 * N_;
    const float* nb = newxyz + (size_t)b * 3 * M_;
#pragma unroll
    for (int c = 0; c < 3; c++)
        X1[(size_t)c*COLS + col] = f2bf(xb[(size_t)c*N_ + n] - nb[(size_t)c*M_ + m]);

    const float* pb = pts + (size_t)b * C_ * N_;
    for (int c = 0; c < C_; c++)
        X1[(size_t)(3+c)*COLS + col] = f2bf(pb[(size_t)c*N_ + n]);

    for (int c = CIN1; c < CP1; c++)
        X1[(size_t)c*COLS + col] = 0;                     // zero channel pad
}

// ---------------- 4) WMMA GEMM + fused input-BN/ReLU + BN-stat epilogue ----------------
// Y[O][COLS] = W[O][CP] x act(X[CP][COLS]); block = 128 thr = 4 waves = 64(O) x 64(col) tile.
// W slab -> LDS via Tensor Data Mover; X tiles -> LDS via async load; WMMA bf16 fp32-acc.
template <int CP, bool ACT>
__global__ void __launch_bounds__(128)
gemm_bn_kernel(const unsigned short* __restrict__ Wb,     // bf16 [Ototal][CP]
               const unsigned short* __restrict__ Xin,    // bf16 [CP][COLS]
               unsigned short* __restrict__ Yout,         // bf16 [Ototal][COLS]
               const float* __restrict__ aScale,
               const float* __restrict__ aShift,
               float* __restrict__ gSum, float* __restrict__ gSq)
{
    __shared__ unsigned short Wlds[64 * CP];    // weight slab, bf16
    __shared__ unsigned short XtLin[32 * 64];   // raw tile, [k][col] (async landing zone)
    __shared__ unsigned short Xt[64 * 32];      // WMMA-transposed tile: [col][k]
    __shared__ float binS[64], binQ[64];

    const int t = threadIdx.x;
    const int wave = t >> 5, lane = t & 31;
    const int hf = lane >> 4, r16 = lane & 15;
    const int oBase = blockIdx.y * 64;
    const int colBase = blockIdx.x * 64;

    if (t < 64) { binS[t] = 0.f; binQ[t] = 0.f; }

    // ---- TDM: one tensor_load_to_lds moves the whole contiguous W slab [64][CP] ----
    if (wave == 0) {
        unsigned long long ga = (unsigned long long)(uintptr_t)(Wb + (size_t)oBase * CP);
        const unsigned elems = 64u * CP;                 // elements of 2 bytes, fits 16b tile_dim0
        u32x4 g0;
        g0[0] = 1u;                                      // count=1, user descriptor
        g0[1] = lds_off(&Wlds[0]);                       // lds_addr
        g0[2] = (unsigned)(ga & 0xFFFFFFFFu);            // global_addr[31:0]
        g0[3] = (unsigned)((ga >> 32) & 0x1FFFFFFu)      // global_addr[56:32]
              | (2u << 30);                              // type=2 ("image")
        u32x8 g1;
        g1[0] = (1u << 16);                              // workgroup_mask=0 | data_size=1 (2B)
        g1[1] = (elems & 0xFFFFu) << 16;                 // tensor_dim0[15:0]
        g1[2] = (elems >> 16) | (1u << 16);              // tensor_dim0[31:16] | tensor_dim1=1
        g1[3] = (elems & 0xFFFFu) << 16;                 // tile_dim0 = elems
        g1[4] = 1u;                                      // tile_dim1=1 | tile_dim2=0
        g1[5] = elems;                                   // tensor_dim0_stride[31:0]
        g1[6] = (elems & 0xFFFFu) << 16;                 // stride hi=0 | dim1_stride[15:0]
        g1[7] = 0u;
        asm volatile("tensor_load_to_lds %0, %1" :: "s"(g0), "s"(g1) : "memory");
        __builtin_amdgcn_s_wait_tensorcnt(0);
    }
    __syncthreads();

    f32x8 acc[4];
    const f32x8 zz = {0.f,0.f,0.f,0.f,0.f,0.f,0.f,0.f};
#pragma unroll
    for (int s = 0; s < 4; s++) acc[s] = zz;

#pragma unroll
    for (int kc = 0; kc < CP; kc += 32) {
        // ---- async stage raw tile [32][64] bf16 -> XtLin (256 x 16B chunks, 2/thread) ----
        {
            const char* gsrc = (const char*)(Xin + (size_t)kc * COLS + colBase);
            unsigned lbase = lds_off(&XtLin[0]);
#pragma unroll
            for (int c = t; c < 256; c += 128) {
                const char* gp = gsrc + (size_t)(c >> 3) * (COLS * 2) + (c & 7) * 16;
                unsigned la = lbase + c * 16;
                asm volatile("global_load_async_to_lds_b128 %0, %1, off"
                             :: "v"(la), "v"(gp) : "memory");
            }
            asm volatile("s_wait_asynccnt 0" ::: "memory");
        }
        __syncthreads();

        // ---- act (prev layer BN+ReLU) + transpose XtLin[k][col] -> Xt[col][k] ----
        {
            const int rr = t >> 2;             // k-local 0..31
            const int cs = (t & 3) * 16;       // col segment 0/16/32/48
            const uint4* lp = (const uint4*)(&XtLin[rr * 64 + cs]);
            uint4 q0 = lp[0], q1 = lp[1];
            float sc = 1.f, sh = 0.f;
            if (ACT) { sc = aScale[kc + rr]; sh = aShift[kc + rr]; }
            const unsigned short* h0 = (const unsigned short*)&q0;
            const unsigned short* h1 = (const unsigned short*)&q1;
#pragma unroll
            for (int j = 0; j < 8; j++) {
                float v = bf2f(h0[j]);
                if (ACT) { v = sc * v + sh; v = v > 0.f ? v : 0.f; }
                Xt[(cs + j) * 32 + rr] = f2bf(v);
            }
#pragma unroll
            for (int j = 0; j < 8; j++) {
                float v = bf2f(h1[j]);
                if (ACT) { v = sc * v + sh; v = v > 0.f ? v : 0.f; }
                Xt[(cs + 8 + j) * 32 + rr] = f2bf(v);
            }
        }
        __syncthreads();

        // ---- A fragment (16x32 bf16, ISA lane layout): two 16B LDS reads ----
        FragBF a;
        {
            int base = (wave * 16 + r16) * CP + kc + 8 * hf;   // ushort units, 16B aligned
            a.q[0] = *(const uint4*)(&Wlds[base]);
            a.q[1] = *(const uint4*)(&Wlds[base + 16]);
        }
#pragma unroll
        for (int s = 0; s < 4; s++) {
            FragBF bf;
            int cb = (s * 16 + r16) * 32 + 8 * hf;             // [col][k], table k order
            bf.q[0] = *(const uint4*)(&Xt[cb]);
            bf.q[1] = *(const uint4*)(&Xt[cb + 16]);
            acc[s] = __builtin_amdgcn_wmma_f32_16x16x32_bf16(
                         false, a.v, false, bf.v, (short)0, acc[s], false, false);
        }
        __syncthreads();
    }

    // epilogue: store bf16 Y + accumulate per-channel BN stats in LDS bins
#pragma unroll
    for (int s = 0; s < 4; s++) {
        int col = colBase + s * 16 + r16;
#pragma unroll
        for (int r = 0; r < 8; r++) {
            int m = r + 8 * hf;
            int oL = wave * 16 + m;
            float v = acc[s][r];
            Yout[(size_t)(oBase + oL) * COLS + col] = f2bf(v);
            atomicAdd(&binS[oL], v);
            atomicAdd(&binQ[oL], v * v);
        }
    }
    __syncthreads();
    if (t < 64) {
        atomicAdd(&gSum[oBase + t], binS[t]);
        atomicAdd(&gSq[oBase + t],  binQ[t]);
    }
}

// ---------------- 5) fold BN stats into per-channel scale/shift ----------------
// relu(g*(y-mu)*rsqrt(var+eps)+be) == relu(scale*y + shift). Conv bias cancels in BN.
__global__ void finalize_kernel(const float* gSum, const float* gSq,
                                const float* g, const float* be,
                                float* sc, float* sh, int O)
{
    int t = blockIdx.x * blockDim.x + threadIdx.x;
    if (t < O) {
        float inv = 1.f / (float)COLS;
        float mu  = gSum[t] * inv;
        float var = gSq[t] * inv - mu * mu;
        float rs  = rsqrtf(var + EPS_);
        float s   = g[t] * rs;
        sc[t] = s;
        sh[t] = be[t] - mu * s;
    }
}

// ---------------- 6) fused BN3+ReLU + max over K ----------------
__global__ void maxpool_kernel(const unsigned short* __restrict__ Y3,
                               const float* __restrict__ sc,
                               const float* __restrict__ sh,
                               float* __restrict__ out)
{
    int idx = blockIdx.x * 256 + threadIdx.x;     // over B*O3*M
    int b = idx / (O3 * M_);
    int rem = idx % (O3 * M_);
    int o = rem / M_, m = rem % M_;
    const uint4* p = (const uint4*)(Y3 + (size_t)o * COLS + (size_t)(b * M_ + m) * K_);
    float s = sc[o], h = sh[o];
    float best = -3.4e38f;
#pragma unroll
    for (int q = 0; q < 4; q++) {                 // 4 x 16B = 32 bf16
        uint4 v4 = p[q];
        const unsigned short* hp = (const unsigned short*)&v4;
#pragma unroll
        for (int k = 0; k < 8; k++) {
            float v = s * bf2f(hp[k]) + h;
            v = v > 0.f ? v : 0.f;
            if (v > best) best = v;
        }
    }
    out[(size_t)B_*3*M_ + (size_t)b*O3*M_ + (size_t)o*M_ + m] = best;
}

// ---------------- 7) prep: weights -> padded bf16, zero stats ----------------
__global__ void prep_kernel(const float* W1, const float* W2, const float* W3,
                            unsigned short* Wb1, unsigned short* Wb2, unsigned short* Wb3,
                            float* statsZero)
{
    int i = blockIdx.x * 256 + threadIdx.x;
    const int t1 = O1 * CP1, t2 = O2 * O1, t3 = O3 * O2;
    if (i < t1) { int o = i / CP1, c = i % CP1;
                  Wb1[i] = (c < CIN1) ? f2bf(W1[o * CIN1 + c]) : (unsigned short)0; }
    int j = i - t1;
    if (j >= 0 && j < t2) Wb2[j] = f2bf(W2[j]);
    int k2 = i - t1 - t2;
    if (k2 >= 0 && k2 < t3) Wb3[k2] = f2bf(W3[k2]);
    if (i < 1024) statsZero[i] = 0.f;   // all stats + scale/shift buffers
}

// ---------------- host: workspace layout + launch pipeline ----------------
extern "C" void kernel_launch(void* const* d_in, const int* in_sizes, int n_in,
                              void* d_out, int out_size, void* d_ws, size_t ws_size,
                              hipStream_t stream)
{
    const float* xyz = (const float*)d_in[0];
    const float* pts = (const float*)d_in[1];
    const float* W1  = (const float*)d_in[2];
    const float* g1  = (const float*)d_in[4];
    const float* be1 = (const float*)d_in[5];
    const float* W2  = (const float*)d_in[6];
    const float* g2  = (const float*)d_in[8];
    const float* be2 = (const float*)d_in[9];
    const float* W3  = (const float*)d_in[10];
    const float* g3  = (const float*)d_in[12];
    const float* be3 = (const float*)d_in[13];
    // b1/b2/b3 (d_in[3,7,11]) are mathematically dead: batch-stat BN cancels conv bias.

    char* ws = (char*)d_ws;
    const size_t OFF_GIDX  = 0;                 // B*M*K int32          =   2 MB
    const size_t OFF_WB1   = 2097152;           // 64*96  bf16
    const size_t OFF_WB2   = 2109440;           // 64*64  bf16
    const size_t OFF_WB3   = 2117632;           // 128*64 bf16
    const size_t OFF_STATS = 2134016;           // 1024 floats
    const size_t OFF_X1    = 4194304;           // 96*COLS bf16         = 100 MB
    const size_t OFF_Y1    = 104857600;         // 64*COLS bf16         =  67 MB
    const size_t OFF_Y2    = OFF_X1;            // aliases X1 (dead after layer 1)
    const size_t OFF_Y3    = 171966464;         // 128*COLS bf16        = 134 MB
    (void)ws_size; (void)in_sizes; (void)n_in; (void)out_size;

    int*            gidx = (int*)(ws + OFF_GIDX);
    unsigned short* Wb1p = (unsigned short*)(ws + OFF_WB1);
    unsigned short* Wb2p = (unsigned short*)(ws + OFF_WB2);
    unsigned short* Wb3p = (unsigned short*)(ws + OFF_WB3);
    float*          st   = (float*)(ws + OFF_STATS);
    float *sum1 = st+0,   *sq1 = st+64,  *sum2 = st+128, *sq2 = st+192;
    float *sum3 = st+256, *sq3 = st+384;
    float *sc1 = st+512, *sh1 = st+576, *sc2 = st+640, *sh2 = st+704;
    float *sc3 = st+768, *sh3 = st+896;
    unsigned short* X1 = (unsigned short*)(ws + OFF_X1);
    unsigned short* Y1 = (unsigned short*)(ws + OFF_Y1);
    unsigned short* Y2 = (unsigned short*)(ws + OFF_Y2);
    unsigned short* Y3 = (unsigned short*)(ws + OFF_Y3);
    float* out = (float*)d_out;   // [new_xyz (B,3,M) | new_points (B,128,M)]

    prep_kernel<<<72, 256, 0, stream>>>(W1, W2, W3, Wb1p, Wb2p, Wb3p, st);
    fps_kernel<<<B_, 256, 0, stream>>>(xyz, out);
    ballq_kernel<<<(B_*M_)/8, 256, 0, stream>>>(xyz, out, gidx);
    gather_kernel<<<COLS/256, 256, 0, stream>>>(xyz, pts, out, gidx, X1);

    // layer 1: 67(->96) -> 64   (no input activation)
    gemm_bn_kernel<CP1, false><<<dim3(COLS/64, O1/64), 128, 0, stream>>>(
        Wb1p, X1, Y1, sc1, sh1, sum1, sq1);
    finalize_kernel<<<1, 128, 0, stream>>>(sum1, sq1, g1, be1, sc1, sh1, O1);

    // layer 2: 64 -> 64 (BN1+ReLU fused into B-tile staging)
    gemm_bn_kernel<O1, true><<<dim3(COLS/64, O2/64), 128, 0, stream>>>(
        Wb2p, Y1, Y2, sc1, sh1, sum2, sq2);
    finalize_kernel<<<1, 128, 0, stream>>>(sum2, sq2, g2, be2, sc2, sh2, O2);

    // layer 3: 64 -> 128 (BN2+ReLU fused)
    gemm_bn_kernel<O2, true><<<dim3(COLS/64, O3/64), 128, 0, stream>>>(
        Wb3p, Y2, Y3, sc2, sh2, sum3, sq3);
    finalize_kernel<<<1, 128, 0, stream>>>(sum3, sq3, g3, be3, sc3, sh3, O3);

    // BN3+ReLU + max over K -> new_points
    maxpool_kernel<<<(B_*O3*M_)/256, 256, 0, stream>>>(Y3, sc3, sh3, out);
}
